// CrossAttention_7473243095473
// MI455X (gfx1250) — compile-verified, matrix operand order checked by reading, not run on hardware
//
#include <hip/hip_runtime.h>
#include <hip/hip_bf16.h>

typedef unsigned short u16;
typedef unsigned int   u32;
typedef unsigned long long u64;
typedef __attribute__((ext_vector_type(16))) __bf16 v16bf;
typedef __attribute__((ext_vector_type(8)))  float  v8f;
typedef __attribute__((ext_vector_type(4)))  u32    u32x4;
typedef __attribute__((ext_vector_type(8)))  u32    u32x8;

union Frag {
    v16bf v;
    uint4 q[2];
};

__device__ __forceinline__ u16 f2bf(float f) {
    u32 u = __float_as_uint(f);
    u32 r = u + 0x7FFFu + ((u >> 16) & 1u);
    return (u16)(r >> 16);
}

__device__ __forceinline__ v8f wmma_bf16(const Frag& a, const Frag& b, v8f c) {
    return __builtin_amdgcn_wmma_f32_16x16x32_bf16(
        false, a.v, false, b.v, (short)0, c, false, false);
}

// ---------------------------------------------------------------------------
// LayerNorm + cast to bf16. One block per row of 1024 floats. 256 threads x4.
// ---------------------------------------------------------------------------
__global__ __launch_bounds__(256) void ln_cast_kernel(
    const float* __restrict__ X, const float* __restrict__ gamma,
    const float* __restrict__ beta, u16* __restrict__ Y)
{
    const int row = blockIdx.x;
    const int tid = threadIdx.x;
    const float4 xv = ((const float4*)(X + (size_t)row * 1024))[tid];
    float s  = xv.x + xv.y + xv.z + xv.w;
    float s2 = xv.x * xv.x + xv.y * xv.y + xv.z * xv.z + xv.w * xv.w;
    #pragma unroll
    for (int m = 1; m < 32; m <<= 1) {
        s  += __shfl_xor(s,  m, 32);
        s2 += __shfl_xor(s2, m, 32);
    }
    __shared__ float ws1[8], ws2[8];
    const int w = tid >> 5, ln = tid & 31;
    if (ln == 0) { ws1[w] = s; ws2[w] = s2; }
    __syncthreads();
    s = 0.f; s2 = 0.f;
    #pragma unroll
    for (int i = 0; i < 8; ++i) { s += ws1[i]; s2 += ws2[i]; }
    const float mu  = s * (1.0f / 1024.0f);
    const float var = s2 * (1.0f / 1024.0f) - mu * mu;
    const float rs  = rsqrtf(var + 1e-5f);
    const float4 gv = ((const float4*)gamma)[tid];
    const float4 bv = ((const float4*)beta)[tid];
    const float y0 = (xv.x - mu) * rs * gv.x + bv.x;
    const float y1 = (xv.y - mu) * rs * gv.y + bv.y;
    const float y2 = (xv.z - mu) * rs * gv.z + bv.z;
    const float y3 = (xv.w - mu) * rs * gv.w + bv.w;
    uint2 o;
    o.x = (u32)f2bf(y0) | ((u32)f2bf(y1) << 16);
    o.y = (u32)f2bf(y2) | ((u32)f2bf(y3) << 16);
    ((uint2*)(Y + (size_t)row * 1024))[tid] = o;
}

// ---------------------------------------------------------------------------
// Cast a 1024x1024 fp32 weight matrix to bf16. 1024 blocks x 256 thr x 4 elem.
// ---------------------------------------------------------------------------
__global__ __launch_bounds__(256) void cast_w_kernel(
    const float* __restrict__ W, u16* __restrict__ Wb)
{
    const int i = blockIdx.x * 256 + threadIdx.x;
    const float4 v = ((const float4*)W)[i];
    uint2 o;
    o.x = (u32)f2bf(v.x) | ((u32)f2bf(v.y) << 16);
    o.y = (u32)f2bf(v.z) | ((u32)f2bf(v.w) << 16);
    ((uint2*)Wb)[i] = o;
}

// ---------------------------------------------------------------------------
// bf16 GEMM: C[8192,1024] = A[8192,1024] x B[1024,1024].
// Block tile 128x128, K-step 32, 8 waves; each wave owns a 32x64 sub-tile
// (2 A-frags x 4 B-frags -> 8 WMMAs per K-step).
// Double-buffered pipeline: while tile i is consumed by WMMAs, the Tensor
// Data Mover streams A tile i+1 into the other LDS buffer (D# pad feature
// reproduces the [TK+8] padded rows) and the B tile i+1 global loads are in
// flight; transpose-stores happen after the WMMAs; s_wait_tensorcnt + one
// barrier close the iteration.
// mode 0: store bf16 as Q/K layout [B,H,N,64]
// mode 1: store bf16 as V-transposed  [B,H,64,N]
// mode 2: store fp32 row-major + bias (final projection -> d_out)
// ---------------------------------------------------------------------------
#define TM 128
#define TN 128
#define TK 32
#define KSTEPS 32

__global__ __launch_bounds__(256) void gemm128x128_kernel(
    const u16* __restrict__ A, const u16* __restrict__ Bm,
    void* __restrict__ Cout, const float* __restrict__ bias, int mode)
{
    __shared__ u16 As[2][TM][TK + 8];   // 2 x 128 x 40 halves (TDM-padded rows)
    __shared__ u16 Bs[2][TN][TK + 8];   // 2 x 128 x 40 halves, (n,k) transposed

    const int tid  = threadIdx.x;
    const int wv   = tid >> 5;
    const int lane = tid & 31;
    const int half = lane >> 4;
    const int l16  = lane & 15;
    const int wm   = wv & 3;         // 4 m-stripes of 32 rows
    const int wn   = wv >> 2;        // 2 n-stripes of 64 cols
    const int m0   = blockIdx.y * TM;
    const int n0   = blockIdx.x * TN;

    // Tensor DMA descriptor group 1 (K-step invariant).
    u32x8 g1;
    g1[0] = 0x00010000u          // data_size = 2 bytes
          | (1u << 20)           // pad_enable
          | (3u << 22)           // pad_interval: 16 dwords (one 32-half row)
          | (3u << 25);          // pad_amount: 4 dwords (8 halves)
    g1[1] = (1024u & 0xffffu) << 16;  // tensor_dim0 = 1024 (lo16)
    g1[2] = (8192u & 0xffffu) << 16;  // tensor_dim0 hi16 = 0 | tensor_dim1 lo16
    g1[3] = (32u << 16);              // tensor_dim1 hi16 = 0 | tile_dim0 = 32
    g1[4] = 128u;                     // tile_dim1 = 128 rows
    g1[5] = 1024u;                    // tensor_dim0_stride (lo32)
    g1[6] = 0u;
    g1[7] = 0u;

    const u32 ldsA[2] = { (u32)(uintptr_t)&As[0][0][0],
                          (u32)(uintptr_t)&As[1][0][0] };

    auto issue_tdm = [&](int k0, int buf) {
        const u64 ga = (u64)(uintptr_t)(A + (size_t)m0 * 1024 + k0);
        u32x4 g0;
        g0[0] = 1u;                            // count = 1 valid descriptor
        g0[1] = ldsA[buf];                     // lds_addr
        g0[2] = (u32)ga;                       // global_addr lo
        g0[3] = (u32)(ga >> 32) | (2u << 30);  // global_addr hi | type=2
        asm volatile("tensor_load_to_lds %0, %1" :: "s"(g0), "s"(g1) : "memory");
    };

    const int kk = (tid >> 4) * 2;      // even k row within tile
    const int c0 = (tid & 15) * 8;      // 8-column group within tile

    auto load_b = [&](int k0, uint4& ra, uint4& rb) {
        ra = *(const uint4*)&Bm[(size_t)(k0 + kk) * 1024 + n0 + c0];
        rb = *(const uint4*)&Bm[(size_t)(k0 + kk + 1) * 1024 + n0 + c0];
    };
    auto store_b = [&](int buf, uint4 ra, uint4 rb) {
        const u32 a4[4] = { ra.x, ra.y, ra.z, ra.w };
        const u32 b4[4] = { rb.x, rb.y, rb.z, rb.w };
        #pragma unroll
        for (int j = 0; j < 4; ++j) {
            const u32 pe = (a4[j] & 0xffffu) | (b4[j] << 16);
            const u32 po = (a4[j] >> 16) | (b4[j] & 0xffff0000u);
            *(u32*)&Bs[buf][c0 + 2 * j][kk]     = pe;
            *(u32*)&Bs[buf][c0 + 2 * j + 1][kk] = po;
        }
    };

    v8f acc[2][4] = {};

    // ---- prologue: stage tile 0 into buffer 0 ----
    if (wv == 0) issue_tdm(0, 0);
    {
        uint4 ra, rb;
        load_b(0, ra, rb);
        store_b(0, ra, rb);
    }
    __builtin_amdgcn_s_wait_tensorcnt(0);
    __syncthreads();

    for (int i = 0; i < KSTEPS; ++i) {
        const int cur = i & 1;
        const int nxt = cur ^ 1;
        const bool more = (i + 1 < KSTEPS);

        // kick off next tile: TDM for A, global loads for B (stores deferred)
        uint4 ra, rb;
        if (more) {
            if (wv == 0) issue_tdm((i + 1) * TK, nxt);
            load_b((i + 1) * TK, ra, rb);
        }

        // ---- compute tile i: 2 A-frags x 4 B-frags -> 8 WMMAs ----
        Frag a[2];
        #pragma unroll
        for (int f = 0; f < 2; ++f) {
            const int arow = wm * 32 + f * 16 + l16;
            a[f].q[0] = *(const uint4*)&As[cur][arow][8 * half];
            a[f].q[1] = *(const uint4*)&As[cur][arow][16 + 8 * half];
        }
        #pragma unroll
        for (int t = 0; t < 4; ++t) {
            Frag b;
            const int brow = wn * 64 + t * 16 + l16;
            b.q[0] = *(const uint4*)&Bs[cur][brow][8 * half];
            b.q[1] = *(const uint4*)&Bs[cur][brow][16 + 8 * half];
            acc[0][t] = wmma_bf16(a[0], b, acc[0][t]);
            acc[1][t] = wmma_bf16(a[1], b, acc[1][t]);
        }

        // finish staging of next tile, then close the iteration
        if (more) store_b(nxt, ra, rb);
        __builtin_amdgcn_s_wait_tensorcnt(0);
        __syncthreads();
    }

    #pragma unroll
    for (int f = 0; f < 2; ++f) {
        #pragma unroll
        for (int t = 0; t < 4; ++t) {
            #pragma unroll
            for (int r = 0; r < 8; ++r) {
                const int row = m0 + wm * 32 + f * 16 + r + 8 * half;
                const int col = n0 + wn * 64 + t * 16 + l16;
                const float v = acc[f][t][r];
                if (mode == 2) {
                    ((float*)Cout)[(size_t)row * 1024 + col] = v + bias[col];
                } else {
                    const int b = row >> 10, n = row & 1023;
                    const int h = col >> 6,  dd = col & 63;
                    const u16 hv = f2bf(v);
                    if (mode == 0)
                        ((u16*)Cout)[((size_t)(b * 16 + h) * 1024 + n) * 64 + dd] = hv;
                    else
                        ((u16*)Cout)[((size_t)(b * 16 + h) * 64 + dd) * 1024 + n] = hv;
                }
            }
        }
    }
}

// ---------------------------------------------------------------------------
// Flash-attention: block = (b,h,64 query rows), 4 waves x 16-row stripes.
// Per 32-key tile: S = Q K^T (4 WMMAs) -> online softmax -> P via LDS ->
// O += P V (4 WMMAs). Writes bf16 attn-out in [B,Nq,H*64] for final GEMM.
// ---------------------------------------------------------------------------
__global__ __launch_bounds__(128) void attn_kernel(
    const u16* __restrict__ Q, const u16* __restrict__ Kb,
    const u16* __restrict__ Vt, const float* __restrict__ bias,
    u16* __restrict__ Aout)
{
    __shared__ u16 Ps[4][16][TK + 8];

    const int tid  = threadIdx.x;
    const int wv   = tid >> 5;
    const int lane = tid & 31;
    const int half = lane >> 4;
    const int l16  = lane & 15;
    const int b  = blockIdx.z, h = blockIdx.y;
    const int bh = b * 16 + h;
    const int q0 = blockIdx.x * 64 + wv * 16;
    const float scale = 0.125f;   // d=64 -> d^-0.5

    // Q fragments (A layout, K=0..31 and K=32..63), straight from global.
    Frag aQ0, aQ1;
    {
        const u16* qp = Q + ((size_t)bh * 1024 + q0 + l16) * 64;
        aQ0.q[0] = *(const uint4*)(qp + 8 * half);
        aQ0.q[1] = *(const uint4*)(qp + 16 + 8 * half);
        aQ1.q[0] = *(const uint4*)(qp + 32 + 8 * half);
        aQ1.q[1] = *(const uint4*)(qp + 48 + 8 * half);
    }

    float m[8], l[8];
    #pragma unroll
    for (int r = 0; r < 8; ++r) { m[r] = -3.0e38f; l[r] = 0.f; }
    v8f o[4] = {};

    for (int kt = 0; kt < 1024; kt += 32) {
        // --- scores for two 16-key tiles ---
        v8f s0, s1;
        {
            const u16* kp = Kb + ((size_t)bh * 1024 + kt + l16) * 64;
            Frag b0, b1;
            b0.q[0] = *(const uint4*)(kp + 8 * half);
            b0.q[1] = *(const uint4*)(kp + 16 + 8 * half);
            b1.q[0] = *(const uint4*)(kp + 32 + 8 * half);
            b1.q[1] = *(const uint4*)(kp + 48 + 8 * half);
            v8f z = {};
            s0 = wmma_bf16(aQ0, b0, z);
            s0 = wmma_bf16(aQ1, b1, s0);
            kp += 16 * 64;
            b0.q[0] = *(const uint4*)(kp + 8 * half);
            b0.q[1] = *(const uint4*)(kp + 16 + 8 * half);
            b1.q[0] = *(const uint4*)(kp + 32 + 8 * half);
            b1.q[1] = *(const uint4*)(kp + 48 + 8 * half);
            v8f z2 = {};
            s1 = wmma_bf16(aQ0, b0, z2);
            s1 = wmma_bf16(aQ1, b1, s1);
        }

        if (kt + 32 < 1024) {
            __builtin_prefetch(Kb + ((size_t)bh * 1024 + kt + 32 + lane) * 64, 0, 1);
            __builtin_prefetch(Vt + ((size_t)bh * 64 + lane) * 1024 + kt + 32, 0, 1);
        }

        // --- bias + online softmax (rows live in 16-lane groups) ---
        #pragma unroll
        for (int r = 0; r < 8; ++r) {
            const int qr = q0 + r + 8 * half;
            const float bv0 = bias[((size_t)h * 1024 + qr) * 1024 + kt + l16];
            const float bv1 = bias[((size_t)h * 1024 + qr) * 1024 + kt + 16 + l16];
            float v0 = s0[r] * scale + bv0;
            float v1 = s1[r] * scale + bv1;
            float mx = fmaxf(v0, v1);
            #pragma unroll
            for (int d = 1; d < 16; d <<= 1) mx = fmaxf(mx, __shfl_xor(mx, d, 32));
            const float mn   = fmaxf(m[r], mx);
            const float corr = __expf(m[r] - mn);
            const float p0 = __expf(v0 - mn);
            const float p1 = __expf(v1 - mn);
            float rs = p0 + p1;
            #pragma unroll
            for (int d = 1; d < 16; d <<= 1) rs += __shfl_xor(rs, d, 32);
            l[r] = l[r] * corr + rs;
            m[r] = mn;
            o[0][r] *= corr; o[1][r] *= corr; o[2][r] *= corr; o[3][r] *= corr;
            Ps[wv][r + 8 * half][l16]      = f2bf(p0);
            Ps[wv][r + 8 * half][16 + l16] = f2bf(p1);
        }
        __syncthreads();

        // --- P (C layout) -> A layout via LDS, then O += P V ---
        Frag aP;
        aP.q[0] = *(const uint4*)&Ps[wv][l16][8 * half];
        aP.q[1] = *(const uint4*)&Ps[wv][l16][16 + 8 * half];
        #pragma unroll
        for (int t = 0; t < 4; ++t) {
            const u16* vp = Vt + ((size_t)bh * 64 + t * 16 + l16) * 1024 + kt;
            Frag bv;
            bv.q[0] = *(const uint4*)(vp + 8 * half);
            bv.q[1] = *(const uint4*)(vp + 16 + 8 * half);
            o[t] = wmma_bf16(aP, bv, o[t]);
        }
        __syncthreads();
    }

    // epilogue: normalize, write bf16 attn-out [B, Nq, H*64]
    #pragma unroll
    for (int r = 0; r < 8; ++r) {
        const float inv = 1.0f / l[r];
        const int row = q0 + r + 8 * half;
        u16* op = Aout + ((size_t)b * 1024 + row) * 1024 + h * 64;
        #pragma unroll
        for (int t = 0; t < 4; ++t)
            op[t * 16 + l16] = f2bf(o[t][r] * inv);
    }
}

// ---------------------------------------------------------------------------
// Host-side orchestration.
// ---------------------------------------------------------------------------
extern "C" void kernel_launch(void* const* d_in, const int* in_sizes, int n_in,
                              void* d_out, int out_size, void* d_ws, size_t ws_size,
                              hipStream_t stream)
{
    const float* x     = (const float*)d_in[0];
    const float* ctx   = (const float*)d_in[1];
    const float* rpb   = (const float*)d_in[2];
    const float* Wq    = (const float*)d_in[3];
    const float* Wk    = (const float*)d_in[4];
    const float* Wv    = (const float*)d_in[5];
    const float* Wo    = (const float*)d_in[6];
    const float* bo    = (const float*)d_in[7];
    const float* gamma = (const float*)d_in[8];
    const float* beta  = (const float*)d_in[9];
    float* out = (float*)d_out;

    const size_t SZ  = (size_t)8192 * 1024;  // activation matrix, bf16 elems
    const size_t WSZ = (size_t)1024 * 1024;  // weight matrix, bf16 elems

    u16* ws  = (u16*)d_ws;
    u16* xn  = ws;
    u16* cn  = xn + SZ;
    u16* wqb = cn + SZ;
    u16* wkb = wqb + WSZ;
    u16* wvb = wkb + WSZ;
    u16* wob = wvb + WSZ;
    u16* qb  = wob + WSZ;   // [B,H,Nq,64]
    u16* kb  = qb + SZ;     // [B,H,Nk,64]
    u16* vt  = kb + SZ;     // [B,H,64,Nk]
    u16* ao  = vt + SZ;     // [B,Nq,H*64]

    // 1) LayerNorm + bf16 cast of x and context
    ln_cast_kernel<<<8192, 256, 0, stream>>>(x,   gamma, beta, xn);
    ln_cast_kernel<<<8192, 256, 0, stream>>>(ctx, gamma, beta, cn);

    // 2) Weight casts
    cast_w_kernel<<<1024, 256, 0, stream>>>(Wq, wqb);
    cast_w_kernel<<<1024, 256, 0, stream>>>(Wk, wkb);
    cast_w_kernel<<<1024, 256, 0, stream>>>(Wv, wvb);
    cast_w_kernel<<<1024, 256, 0, stream>>>(Wo, wob);

    // 3) QKV projections (WMMA GEMMs)
    const dim3 gdim(1024 / TN, 8192 / TM);   // (8, 64)
    gemm128x128_kernel<<<gdim, 256, 0, stream>>>(xn, wqb, qb, nullptr, 0);
    gemm128x128_kernel<<<gdim, 256, 0, stream>>>(cn, wkb, kb, nullptr, 0);
    gemm128x128_kernel<<<gdim, 256, 0, stream>>>(cn, wvb, vt, nullptr, 1);

    // 4) Fused attention (scores + bias + softmax + PV)
    attn_kernel<<<dim3(16, 16, 8), 128, 0, stream>>>(qb, kb, vt, rpb, ao);

    // 5) Output projection + bias -> d_out (fp32)
    gemm128x128_kernel<<<gdim, 256, 0, stream>>>(ao, wob, out, bo, 2);
}
